// RBFN_16819091931602
// MI455X (gfx1250) — matrix operand order repaired and checked by reading, not compiled, not run on hardware
//
#include <hip/hip_runtime.h>

#define N_INPUT     16384
#define NUM_CENTERS 2048
#define DIM         256
#define N_OUT       64
#define W_K         (NUM_CENTERS + DIM)   // 2304

#define BLOCK_M   64
#define NTHREADS  128
#define CCHUNK    32

// LDS strides (in bf16 elements); chosen so row offsets are 16B-aligned for
// ds_load_b128 and so the 16-lane fragment reads spread across banks.
#define XSTR 264   // 256 + 8  (row offset 528 B = 33*16)
#define CSTR 264
#define WSTR 48    // 32 + 16  (row offset 96 B = 6*16)
#define RSTR 48

typedef __attribute__((ext_vector_type(4)))  float        f32x4;
typedef __attribute__((ext_vector_type(2)))  unsigned int u32x2;
typedef __attribute__((ext_vector_type(4)))  unsigned int u32x4;
typedef __attribute__((ext_vector_type(8)))  float        v8f;
typedef __attribute__((ext_vector_type(16))) __bf16       v16bf;
typedef __attribute__((ext_vector_type(2)))  __bf16       v2bf;

union FragBF {
    u32x4 q[2];
    v16bf v;
};

// f32 -> bf16 via compiler fptrunc: lets the backend choose the best gfx1250
// lowering (hardware cvt if present; RNE emulation otherwise).
static __device__ __forceinline__ unsigned int pk(float lo, float hi) {
    union { v2bf v; unsigned int u; } t;
    t.v[0] = (__bf16)lo;
    t.v[1] = (__bf16)hi;
    return t.u;
}
static __device__ __forceinline__ unsigned short f2bf(float f) {
    union { __bf16 v; unsigned short u; } t;
    t.v = (__bf16)f;
    return t.u;
}

#define WMMA_BF16(A, B, C) \
    __builtin_amdgcn_wmma_f32_16x16x32_bf16(false, (A), false, (B), (short)0, (C), false, false)

__global__ __launch_bounds__(NTHREADS)
void rbfn_fused(const float* __restrict__ batches,
                const float* __restrict__ centers,
                const float* __restrict__ beta,
                const float* __restrict__ lin_w,
                const float* __restrict__ lin_b,
                float* __restrict__ out)
{
    __shared__ __align__(16) unsigned short sX[BLOCK_M * XSTR];   // 33 KB
    __shared__ __align__(16) unsigned short sC[CCHUNK * CSTR];    // 16.5 KB
    __shared__ __align__(16) unsigned short sW[N_OUT * WSTR];     // 6 KB
    __shared__ __align__(16) unsigned short sR[4 * 16 * RSTR];    // 6 KB (per-wave radial staging)
    __shared__ float sXsq[BLOCK_M];
    __shared__ float sCsq[CCHUNK];
    __shared__ float sBeta[CCHUNK];
    __shared__ float sPart[NTHREADS];

    const int tid  = threadIdx.x;
    const int lane = tid & 31;          // wave32
    const int wave = tid >> 5;          // 4 waves
    const int m16  = lane & 15;
    const int h    = lane >> 4;         // lane half (K-phase select per WMMA layout)
    const int rowbase = blockIdx.x * BLOCK_M;

    // ---------------- stage X tile (f32 -> bf16 LDS) fused with |x|^2 ----------------
    // 2 threads per row, 32 consecutive f32x4 each; accumulate sum of squares inline.
    {
        int r  = tid >> 1;
        int hh = tid & 1;
        const float* src = batches + (size_t)(rowbase + r) * DIM + hh * 128;
        unsigned short* dst = &sX[r * XSTR + hh * 128];
        float s = 0.f;
        #pragma unroll
        for (int i = 0; i < 32; ++i) {
            f32x4 v = *(const f32x4*)(src + i * 4);
            s += v.x * v.x + v.y * v.y + v.z * v.z + v.w * v.w;
            u32x2 w; w.x = pk(v.x, v.y); w.y = pk(v.z, v.w);
            *(u32x2*)(dst + i * 4) = w;
        }
        sPart[tid] = s;
    }
    __syncthreads();
    if (tid < BLOCK_M) sXsq[tid] = sPart[2 * tid] + sPart[2 * tid + 1];

    // ---- hoist this wave's X A-fragments (16 rows x 256 K) into registers ----
    // 8 k-chunks x 32B/lane = 64 VGPRs; eliminates all repeated A-side LDS reads.
    const int arowOff = (wave * 16 + m16) * XSTR;
    FragBF XA[DIM / 32];
    #pragma unroll
    for (int kc = 0; kc < DIM / 32; ++kc) {
        XA[kc].q[0] = *(const u32x4*)&sX[arowOff + kc * 32 + 16 * h];
        XA[kc].q[1] = *(const u32x4*)&sX[arowOff + kc * 32 + 16 * h + 8];
    }

    v8f acc[4];
    #pragma unroll
    for (int t = 0; t < 4; ++t) acc[t] = (v8f){};

    // ---------------- phase 1: out += X @ W1^T  (K = DIM) ----------------
    for (int kc = 0; kc < DIM / 32; ++kc) {
        const int kb = kc * 32;
        __syncthreads();   // protect sW reuse
        for (int idx = tid; idx < N_OUT * 8; idx += NTHREADS) {   // 64 rows x 8 f32x4
            int n  = idx >> 3;
            int c4 = idx & 7;
            f32x4 v = *(const f32x4*)(lin_w + (size_t)n * W_K + kb + c4 * 4);
            u32x2 w; w.x = pk(v.x, v.y); w.y = pk(v.z, v.w);
            *(u32x2*)&sW[n * WSTR + c4 * 4] = w;
        }
        __syncthreads();
        #pragma unroll
        for (int t = 0; t < 4; ++t) {
            FragBF B;
            B.q[0] = *(const u32x4*)&sW[(16 * t + m16) * WSTR + 16 * h];
            B.q[1] = *(const u32x4*)&sW[(16 * t + m16) * WSTR + 16 * h + 8];
            acc[t] = WMMA_BF16(XA[kc].v, B.v, acc[t]);
        }
    }

    // x_sq values for this lane's C-tile rows are loop-invariant: hoist once.
    float xs[8];
    #pragma unroll
    for (int r = 0; r < 8; ++r) xs[r] = sXsq[wave * 16 + r + 8 * h];

    // ---------------- fused center loop ----------------
    for (int cc = 0; cc < NUM_CENTERS / CCHUNK; ++cc) {
        const int cb = cc * CCHUNK;
        __syncthreads();   // protect sC / sW reuse from previous iteration

        // stage centers chunk (f32 -> bf16) fused with |c|^2 partials:
        // 4 threads per row, 16 consecutive f32x4 each.
        {
            int r = tid >> 2, q = tid & 3;
            const float* src = centers + (size_t)(cb + r) * DIM + q * 64;
            unsigned short* dst = &sC[r * CSTR + q * 64];
            float s = 0.f;
            #pragma unroll
            for (int i = 0; i < 16; ++i) {
                f32x4 v = *(const f32x4*)(src + i * 4);
                s += v.x * v.x + v.y * v.y + v.z * v.z + v.w * v.w;
                u32x2 w; w.x = pk(v.x, v.y); w.y = pk(v.z, v.w);
                *(u32x2*)(dst + i * 4) = w;
            }
            sPart[tid] = s;
        }
        if (tid < CCHUNK) sBeta[tid] = beta[cb + tid];
        // stage W2 chunk (radial part of lin_w)
        for (int idx = tid; idx < N_OUT * 8; idx += NTHREADS) {
            int n  = idx >> 3;
            int c4 = idx & 7;
            f32x4 v = *(const f32x4*)(lin_w + (size_t)n * W_K + DIM + cb + c4 * 4);
            u32x2 w; w.x = pk(v.x, v.y); w.y = pk(v.z, v.w);
            *(u32x2*)&sW[n * WSTR + c4 * 4] = w;
        }
        __syncthreads();
        if (tid < CCHUNK)
            sCsq[tid] = sPart[4 * tid] + sPart[4 * tid + 1] + sPart[4 * tid + 2] + sPart[4 * tid + 3];
        __syncthreads();

        // prefetch next centers chunk toward L2/WGP$
        if (cc + 1 < NUM_CENTERS / CCHUNK)
            __builtin_prefetch(centers + (size_t)(cb + CCHUNK) * DIM + tid * 16, 0, 1);

        // ---- cross GEMM: [16 x 32] tile of X @ C^T per wave (A from registers) ----
        v8f x0 = (v8f){}, x1 = (v8f){};
        #pragma unroll
        for (int kc = 0; kc < DIM / 32; ++kc) {
            const int kb = kc * 32;
            FragBF B0, B1;
            B0.q[0] = *(const u32x4*)&sC[m16 * CSTR + kb + 16 * h];
            B0.q[1] = *(const u32x4*)&sC[m16 * CSTR + kb + 16 * h + 8];
            B1.q[0] = *(const u32x4*)&sC[(16 + m16) * CSTR + kb + 16 * h];
            B1.q[1] = *(const u32x4*)&sC[(16 + m16) * CSTR + kb + 16 * h + 8];
            x0 = WMMA_BF16(XA[kc].v, B0.v, x0);
            x1 = WMMA_BF16(XA[kc].v, B1.v, x1);
        }

        // ---- elementwise: radial = exp(-beta * (|x|^2 + |c|^2 - 2*cross)) ----
        const float cs0 = sCsq[m16],  cs1 = sCsq[16 + m16];
        const float bt0 = sBeta[m16], bt1 = sBeta[16 + m16];
        unsigned short* rw = &sR[wave * 16 * RSTR];
        #pragma unroll
        for (int r = 0; r < 8; ++r) {
            float d0 = xs[r] + cs0 - 2.0f * x0[r];
            float d1 = xs[r] + cs1 - 2.0f * x1[r];
            rw[(r + 8 * h) * RSTR + m16]      = f2bf(__expf(-bt0 * d0));
            rw[(r + 8 * h) * RSTR + 16 + m16] = f2bf(__expf(-bt1 * d1));
        }
        __asm__ volatile("s_wait_dscnt 0x0" ::: "memory");   // per-wave LDS RAW fence

        // ---- second GEMM: out += radial[16x32] @ W2chunk^T ----
        FragBF RA;
        RA.q[0] = *(const u32x4*)&rw[m16 * RSTR + 16 * h];
        RA.q[1] = *(const u32x4*)&rw[m16 * RSTR + 16 * h + 8];
        #pragma unroll
        for (int t = 0; t < 4; ++t) {
            FragBF B;
            B.q[0] = *(const u32x4*)&sW[(16 * t + m16) * WSTR + 16 * h];
            B.q[1] = *(const u32x4*)&sW[(16 * t + m16) * WSTR + 16 * h + 8];
            acc[t] = WMMA_BF16(RA.v, B.v, acc[t]);
        }
    }

    // ---------------- epilogue: bias + store (C layout: row = r + 8h, col = 16t + m16) ----
    #pragma unroll
    for (int t = 0; t < 4; ++t) {
        float bias = lin_b[16 * t + m16];
        #pragma unroll
        for (int r = 0; r < 8; ++r) {
            int m = rowbase + wave * 16 + r + 8 * h;
            out[(size_t)m * N_OUT + 16 * t + m16] = acc[t][r] + bias;
        }
    }
}

extern "C" void kernel_launch(void* const* d_in, const int* in_sizes, int n_in,
                              void* d_out, int out_size, void* d_ws, size_t ws_size,
                              hipStream_t stream) {
    (void)in_sizes; (void)n_in; (void)d_ws; (void)ws_size; (void)out_size;
    const float* batches = (const float*)d_in[0];
    const float* centers = (const float*)d_in[1];
    const float* beta    = (const float*)d_in[2];
    const float* lin_w   = (const float*)d_in[3];
    const float* lin_b   = (const float*)d_in[4];
    float* out = (float*)d_out;

    dim3 grid(N_INPUT / BLOCK_M);   // 256 blocks
    dim3 block(NTHREADS);           // 128 threads = 4 wave32 waves
    hipLaunchKernelGGL(rbfn_fused, grid, block, 0, stream,
                       batches, centers, beta, lin_w, lin_b, out);
}
